// CrossAttention_90494960927146
// MI455X (gfx1250) — compile-verified, matrix operand order checked by reading, not run on hardware
//
#include <hip/hip_runtime.h>
#include <hip/hip_bf16.h>
#include <cstdint>
#include <cstddef>

// ---------------------------------------------------------------------------
// Problem constants (reference: B=128, C=128, H=W=32)
// ---------------------------------------------------------------------------
#define B_   128
#define C_   128
#define HW_  1024
#define NE_  ((size_t)B_ * C_ * HW_)   // 16,777,216 elements per [B,C,H,W]
#define EPS_ 1e-5f

typedef __attribute__((ext_vector_type(16))) _Float16 v16h;
typedef __attribute__((ext_vector_type(8)))  float    v8f;

// Padded per-lane fragment stride in LDS: 24 halves = 48B = 12 DWORDs.
// (16B-aligned for ds_load_b128; 12 DWORDs/lane -> only 2-way bank conflict
//  vs 4-way at the natural 8-DWORD stride.)
#define FRAG_STRIDE 24

// ---------------------------------------------------------------------------
// WMMA layout (wave32, v_wmma_f32_16x16x32_f16), per ISA 7.12.2:
//   A (16x32): lane L<16 -> M=L, K in {0..7}U{16..23}; L>=16 -> M=L-16,
//              K in {8..15}U{24..31}
//   B (32x16): lane -> N=L&15, same K chunking.
//   C/D (16x16 f32): VGPR r -> rows {r, r+8} by lane half, N = lane&15.
// Fragment slot for K-index k: laneHalf=(k>>3)&1, e=(k&7)|((k>>4)<<3)
// ---------------------------------------------------------------------------
__device__ __forceinline__ v8f wmma_f16(v16h a, v16h b, v8f c) {
  return __builtin_amdgcn_wmma_f32_16x16x32_f16(false, a, false, b, (short)0, c,
                                                false, false);
}

// LDS slot (padded) for B-matrix element (k, n), n in 0..15.
__device__ __forceinline__ int bfrag_slot(int k, int n) {
  int lane = (((k >> 3) & 1) << 4) | n;
  int e    = (k & 7) | ((k >> 4) << 3);
  return lane * FRAG_STRIDE + e;
}

// A-fragment from row-major LDS tile (rows contiguous: 2 x ds_load_b128).
__device__ __forceinline__ v16h load_a_rowmajor_lds(const _Float16* a, int lda) {
  const int lane = threadIdx.x & 31;
  const _Float16* r = a + (size_t)(lane & 15) * lda + ((lane >> 4) << 3);
  v16h f;
#pragma unroll
  for (int i = 0; i < 8; ++i) { f[i] = r[i]; f[8 + i] = r[16 + i]; }
  return f;
}

// ---------------------------------------------------------------------------
// f32 -> f16 conversion (grid-stride) for activations.
// ---------------------------------------------------------------------------
__global__ __launch_bounds__(256) void cvt_f32_f16(const float* __restrict__ src,
                                                   _Float16* __restrict__ dst,
                                                   size_t n) {
  size_t i = (size_t)blockIdx.x * blockDim.x + threadIdx.x;
  size_t stride = (size_t)gridDim.x * blockDim.x;
  for (; i < n; i += stride) dst[i] = (_Float16)src[i];
}

// ---------------------------------------------------------------------------
// Weight pre-pack: f32 row-major [M][K] -> f16 fragment-major
//   packed[(((mt*KSTEPS + kt)*32 + lane)*16 + e)]
// so an A-fragment is one contiguous 32B per-lane (coalesced) load.
// ---------------------------------------------------------------------------
__global__ __launch_bounds__(256) void pack_a(const float* __restrict__ w,
                                              _Float16* __restrict__ out,
                                              int K, size_t total) {
  size_t i = (size_t)blockIdx.x * blockDim.x + threadIdx.x;
  if (i >= total) return;
  const int ksteps = K >> 5;
  int e    = (int)(i & 15);
  int lane = (int)((i >> 4) & 31);
  size_t t = i >> 9;
  int kt = (int)(t % ksteps);
  int mt = (int)(t / ksteps);
  int m  = mt * 16 + (lane & 15);
  int k0 = (lane >> 4) << 3;
  int k  = kt * 32 + ((e < 8) ? (k0 + e) : (k0 + 16 + (e - 8)));
  out[i] = (_Float16)w[(size_t)m * K + k];
}

// ---------------------------------------------------------------------------
// 1x1 projections: out[b] = W(128x128) @ X[b](128x1024) + bias, f16 out.
// grid = (32 n-tiles, B, 6 projections), block = 256 (8 waves).
// Block computes 128 x 32; each wave: one 16-row tile x two 16-col tiles
// (2 WMMAs per A-fragment). X 32x32 tile staged in LDS in padded frag order.
// ---------------------------------------------------------------------------
struct ProjBias { const float* b[6]; };

__global__ __launch_bounds__(256) void proj_gemm(const _Float16* __restrict__ st_h,
                                                 const _Float16* __restrict__ phy_h,
                                                 const _Float16* __restrict__ wpack, // 6 x 128x128 frag-packed
                                                 ProjBias pb,
                                                 _Float16* __restrict__ qkv) {       // 6 x NE_
  __shared__ alignas(32) _Float16 sB[2 * 32 * FRAG_STRIDE];
  const int nt    = blockIdx.x;            // 0..31
  const int b     = blockIdx.y;            // 0..127
  const int pz    = blockIdx.z;            // 0..5: q_st,k_st,v_st,q_phy,k_phy,v_phy
  const int tid   = threadIdx.x;
  const int wv    = tid >> 5;
  const int lane  = tid & 31;
  const int nbase = nt * 32;

  const _Float16* X  = ((pz < 3) ? st_h : phy_h) + (size_t)b * (C_ * HW_);
  const _Float16* Wp = wpack + (size_t)pz * (C_ * C_);
  _Float16*      out = qkv + (size_t)pz * NE_ + (size_t)b * (C_ * HW_);

  v8f acc0 = {}, acc1 = {};
#pragma unroll 1
  for (int kt = 0; kt < 4; ++kt) {
#pragma unroll
    for (int u = 0; u < 4; ++u) {          // 1024 elements, 4 per thread
      int idx = tid + u * 256;
      int kr = idx >> 5, nc = idx & 31;
      int ni = nc >> 4, nn = nc & 15;
      sB[ni * (32 * FRAG_STRIDE) + bfrag_slot(kr, nn)] =
          X[(size_t)(kt * 32 + kr) * HW_ + nbase + nc];
    }
    __syncthreads();
    v16h a   = *(const v16h*)(Wp + ((size_t)(wv * 4 + kt) * 32 + lane) * 16);
    v16h bb0 = *(const v16h*)(sB + lane * FRAG_STRIDE);
    v16h bb1 = *(const v16h*)(sB + 32 * FRAG_STRIDE + lane * FRAG_STRIDE);
    acc0 = wmma_f16(a, bb0, acc0);
    acc1 = wmma_f16(a, bb1, acc1);
    __syncthreads();
  }
  const float* bias = pb.b[pz];
  const int n = lane & 15, mb = (lane >> 4) << 3;
#pragma unroll
  for (int r = 0; r < 8; ++r) {
    int m = wv * 16 + mb + r;
    float bi = bias[m];
    out[(size_t)m * HW_ + nbase + n]      = (_Float16)(acc0[r] + bi);
    out[(size_t)m * HW_ + nbase + 16 + n] = (_Float16)(acc1[r] + bi);
  }
}

// ---------------------------------------------------------------------------
// Attention: per (b,c): S = softmax(Q@K); O = softmax(S@V) [+ residual]
// grid = (B*C, 4 variants), block = 128 (4 waves: 2x2 tiles of the 32x32).
// K/V staged in LDS in padded frag order; Q/P row-major for the A side.
// Writes f16 directly into concat [B,512,H,W].
// ---------------------------------------------------------------------------
__global__ __launch_bounds__(128) void attn_kernel(const _Float16* __restrict__ qkv,
                                                   const float* __restrict__ st,
                                                   const float* __restrict__ phy,
                                                   _Float16* __restrict__ concat) {
  __shared__ alignas(32) _Float16 sQ[1024];                    // row-major [32][32]
  __shared__ alignas(32) _Float16 sKf[2 * 32 * FRAG_STRIDE];   // padded frag order
  __shared__ alignas(32) _Float16 sVf[2 * 32 * FRAG_STRIDE];
  __shared__ alignas(32) _Float16 sP[1024];                    // row-major
  __shared__ float sS[1024];

  const int bc  = blockIdx.x;
  const int b   = bc >> 7;
  const int c   = bc & 127;
  const int var = blockIdx.y;              // 0:st 1:phy 2:st_phy 3:phy_st
  const int tid = threadIdx.x;
  const int wv  = tid >> 5;

  const int qi = (var == 0) ? 0 : (var == 1) ? 3 : (var == 2) ? 3 : 0;
  const int ki = (var == 0) ? 1 : (var == 1) ? 4 : (var == 2) ? 1 : 4;
  const int vi = (var == 0) ? 2 : (var == 1) ? 5 : (var == 2) ? 2 : 5;
  const float* res = (var == 0) ? st : (var == 1) ? phy : nullptr;

  const size_t off = (size_t)b * (C_ * HW_) + (size_t)c * HW_;
  const _Float16* Q = qkv + (size_t)qi * NE_ + off;
  const _Float16* K = qkv + (size_t)ki * NE_ + off;
  const _Float16* V = qkv + (size_t)vi * NE_ + off;

#pragma unroll
  for (int u = 0; u < 8; ++u) {            // 1024 elements, 8 per thread
    int i = tid + u * 128;
    int kr = i >> 5, n = i & 31;
    int ni = n >> 4, nn = n & 15;
    int slot = ni * (32 * FRAG_STRIDE) + bfrag_slot(kr, nn);
    sQ[i]     = Q[i];
    sKf[slot] = K[i];
    sVf[slot] = V[i];
  }
  __syncthreads();

  const int mi = wv >> 1, ni = wv & 1;
  const int lane = tid & 31, n = lane & 15, mb = (lane >> 4) << 3;

  { // S = Q @ K   (K dim = 32 -> single WMMA per 16x16 tile)
    v16h a  = load_a_rowmajor_lds(sQ + mi * 16 * 32, 32);
    v16h bb = *(const v16h*)(sKf + ni * (32 * FRAG_STRIDE) + lane * FRAG_STRIDE);
    v8f d = {};
    d = wmma_f16(a, bb, d);
#pragma unroll
    for (int r = 0; r < 8; ++r) sS[(mi * 16 + mb + r) * 32 + ni * 16 + n] = d[r];
  }
  __syncthreads();
  if (tid < 32) { // row softmax -> f16 P
    float mx = -1e30f;
    for (int j = 0; j < 32; ++j) mx = fmaxf(mx, sS[tid * 32 + j]);
    float sum = 0.f, e[32];
    for (int j = 0; j < 32; ++j) { e[j] = __expf(sS[tid * 32 + j] - mx); sum += e[j]; }
    float inv = 1.f / sum;
    for (int j = 0; j < 32; ++j) sP[tid * 32 + j] = (_Float16)(e[j] * inv);
  }
  __syncthreads();
  { // O_pre = P @ V
    v16h a  = load_a_rowmajor_lds(sP + mi * 16 * 32, 32);
    v16h bb = *(const v16h*)(sVf + ni * (32 * FRAG_STRIDE) + lane * FRAG_STRIDE);
    v8f d = {};
    d = wmma_f16(a, bb, d);
#pragma unroll
    for (int r = 0; r < 8; ++r) sS[(mi * 16 + mb + r) * 32 + ni * 16 + n] = d[r];
  }
  __syncthreads();
  if (tid < 32) { // second row softmax, residual, write concat channel block
    float mx = -1e30f;
    for (int j = 0; j < 32; ++j) mx = fmaxf(mx, sS[tid * 32 + j]);
    float sum = 0.f, e[32];
    for (int j = 0; j < 32; ++j) { e[j] = __expf(sS[tid * 32 + j] - mx); sum += e[j]; }
    float inv = 1.f / sum;
    _Float16* dst = concat + ((size_t)b * 512 + var * 128 + c) * HW_ + tid * 32;
    const float* rs = res ? (res + off + tid * 32) : nullptr;
    for (int j = 0; j < 32; ++j) {
      float v = e[j] * inv;
      if (rs) v += rs[j];
      dst[j] = (_Float16)v;
    }
  }
}

// ---------------------------------------------------------------------------
// Implicit-GEMM conv over concat [B,512,32,32] -> out [B,128,32,32] (f32).
// grid = (32 n-tiles, B), block = 256 (8 waves = all 128 Cout).
// Block computes 128 x 32: each wave does 2 WMMAs per K-step on one shared
// im2col tile (padded frag order in LDS). Templated on KK/PAD so index math
// is shifts/const-mul. Weights are frag-packed in global memory.
// ---------------------------------------------------------------------------
template <int KK, int PAD>
__global__ __launch_bounds__(256) void conv_gemm(const _Float16* __restrict__ concat,
                                                 const _Float16* __restrict__ wpack, // [8][KSTEPS][32][16]
                                                 const float* __restrict__ bias,     // nullable
                                                 float* __restrict__ out) {          // [B][128][1024]
  constexpr int KK2    = KK * KK;
  constexpr int K      = 512 * KK2;
  constexpr int KSTEPS = K / 32;
  __shared__ alignas(32) _Float16 sB[2 * 32 * FRAG_STRIDE];

  const int nt = blockIdx.x, b = blockIdx.y;
  const int tid = threadIdx.x, wv = tid >> 5, lane = tid & 31;
  const int nbase = nt * 32;

  v8f acc0 = {}, acc1 = {};
#pragma unroll 1
  for (int kt = 0; kt < KSTEPS; ++kt) {
#pragma unroll
    for (int u = 0; u < 4; ++u) {          // 1024 elements, 4 per thread
      int idx = tid + u * 256;
      int kr = idx >> 5, nc = idx & 31;
      int kg  = kt * 32 + kr;
      int ci  = kg / KK2;                  // const divisor
      int rem = kg - ci * KK2;
      int kh  = rem / KK;
      int kw  = rem - kh * KK;
      int p = nbase + nc;
      int h = p >> 5, w = p & 31;
      int ih = h + kh - PAD, iw = w + kw - PAD;
      _Float16 v = (_Float16)0.f;
      if ((unsigned)ih < 32u && (unsigned)iw < 32u)
        v = concat[((size_t)b * 512 + ci) * HW_ + ih * 32 + iw];
      int ni = nc >> 4, nn = nc & 15;
      sB[ni * (32 * FRAG_STRIDE) + bfrag_slot(kr, nn)] = v;
    }
    __syncthreads();
    v16h a   = *(const v16h*)(wpack + ((size_t)(wv * KSTEPS + kt) * 32 + lane) * 16);
    v16h bb0 = *(const v16h*)(sB + lane * FRAG_STRIDE);
    v16h bb1 = *(const v16h*)(sB + 32 * FRAG_STRIDE + lane * FRAG_STRIDE);
    acc0 = wmma_f16(a, bb0, acc0);
    acc1 = wmma_f16(a, bb1, acc1);
    __syncthreads();
  }
  const int n = lane & 15, mb = (lane >> 4) << 3;
#pragma unroll
  for (int r = 0; r < 8; ++r) {
    int m = wv * 16 + mb + r;
    float bi = bias ? bias[m] : 0.f;
    out[((size_t)b * 128 + m) * HW_ + nbase + n]      = acc0[r] + bi;
    out[((size_t)b * 128 + m) * HW_ + nbase + 16 + n] = acc1[r] + bi;
  }
}

// ---------------------------------------------------------------------------
// Per-batch LayerNorm stats over C*H*W = 131072 elements.
// ---------------------------------------------------------------------------
__global__ __launch_bounds__(256) void ln_stats(const float* __restrict__ x,
                                                float* __restrict__ stats) { // [B][2]
  __shared__ float ssum[256], ssq[256];
  const int b = blockIdx.x;
  const float* p = x + (size_t)b * (C_ * HW_);
  float s = 0.f, q = 0.f;
  for (int i = threadIdx.x; i < C_ * HW_; i += 256) { float v = p[i]; s += v; q += v * v; }
  ssum[threadIdx.x] = s; ssq[threadIdx.x] = q;
  __syncthreads();
  for (int st = 128; st > 0; st >>= 1) {
    if ((int)threadIdx.x < st) {
      ssum[threadIdx.x] += ssum[threadIdx.x + st];
      ssq[threadIdx.x]  += ssq[threadIdx.x + st];
    }
    __syncthreads();
  }
  if (threadIdx.x == 0) {
    const float inv = 1.f / (float)(C_ * HW_);
    float mean = ssum[0] * inv;
    stats[b * 2]     = mean;
    stats[b * 2 + 1] = ssq[0] * inv - mean * mean;
  }
}

// ---------------------------------------------------------------------------
// Final: sigmoid((LN(x1)+LN(x2)+LN(x3))/3) + LN(xc)
// ---------------------------------------------------------------------------
__global__ __launch_bounds__(256) void final_combine(
    const float* __restrict__ x1, const float* __restrict__ x2,
    const float* __restrict__ x3, const float* __restrict__ xc,
    const float* __restrict__ stats,   // [4][B][2]
    const float* __restrict__ g3, const float* __restrict__ be3,
    const float* __restrict__ g5, const float* __restrict__ be5,
    const float* __restrict__ g7, const float* __restrict__ be7,
    const float* __restrict__ g1, const float* __restrict__ be1,
    float* __restrict__ out) {
  size_t i = (size_t)blockIdx.x * blockDim.x + threadIdx.x;
  if (i >= NE_) return;
  int b    = (int)(i >> 17);        // / 131072
  int cidx = (int)(i & 131071);     // c*HW + hw, matches g/be [C,H,W] layout
  float m1 = stats[(0 * B_ + b) * 2], v1 = stats[(0 * B_ + b) * 2 + 1];
  float m2 = stats[(1 * B_ + b) * 2], v2 = stats[(1 * B_ + b) * 2 + 1];
  float m3 = stats[(2 * B_ + b) * 2], v3 = stats[(2 * B_ + b) * 2 + 1];
  float m4 = stats[(3 * B_ + b) * 2], v4 = stats[(3 * B_ + b) * 2 + 1];
  float a1 = (x1[i] - m1) * rsqrtf(v1 + EPS_) * g3[cidx] + be3[cidx];
  float a2 = (x2[i] - m2) * rsqrtf(v2 + EPS_) * g5[cidx] + be5[cidx];
  float a3 = (x3[i] - m3) * rsqrtf(v3 + EPS_) * g7[cidx] + be7[cidx];
  float sgm = 1.f / (1.f + __expf(-(a1 + a2 + a3) * (1.f / 3.f)));
  float at  = (xc[i] - m4) * rsqrtf(v4 + EPS_) * g1[cidx] + be1[cidx];
  out[i] = sgm + at;
}

// ---------------------------------------------------------------------------
// Launch
// ---------------------------------------------------------------------------
extern "C" void kernel_launch(void* const* d_in, const int* in_sizes, int n_in,
                              void* d_out, int out_size, void* d_ws, size_t ws_size,
                              hipStream_t stream) {
  (void)in_sizes; (void)n_in; (void)out_size; (void)ws_size;

  // Inputs in setup_inputs() dict order.
  const float* st  = (const float*)d_in[0];
  const float* phy = (const float*)d_in[1];
  const float* wproj[6] = { (const float*)d_in[2],  (const float*)d_in[4],
                            (const float*)d_in[6],  (const float*)d_in[8],
                            (const float*)d_in[10], (const float*)d_in[12] };
  ProjBias pb;
  pb.b[0] = (const float*)d_in[3];  pb.b[1] = (const float*)d_in[5];
  pb.b[2] = (const float*)d_in[7];  pb.b[3] = (const float*)d_in[9];
  pb.b[4] = (const float*)d_in[11]; pb.b[5] = (const float*)d_in[13];
  const float* w3 = (const float*)d_in[14];
  const float* g3 = (const float*)d_in[15]; const float* be3 = (const float*)d_in[16];
  const float* w5 = (const float*)d_in[17];
  const float* g5 = (const float*)d_in[18]; const float* be5 = (const float*)d_in[19];
  const float* w7 = (const float*)d_in[20];
  const float* g7 = (const float*)d_in[21]; const float* be7 = (const float*)d_in[22];
  const float* w1 = (const float*)d_in[23];
  const float* g1 = (const float*)d_in[24]; const float* be1 = (const float*)d_in[25];
  const float* b1 = (const float*)d_in[26];

  // Workspace carve-out (256B aligned slices).
  char* ws = (char*)d_ws;
  size_t o = 0;
  auto take = [&](size_t bytes) -> char* {
    char* p = ws + o;
    o += (bytes + 255) & ~(size_t)255;
    return p;
  };
  _Float16* st_h    = (_Float16*)take(NE_ * 2);
  _Float16* phy_h   = (_Float16*)take(NE_ * 2);
  _Float16* qkv     = (_Float16*)take(6 * NE_ * 2);
  _Float16* concat  = (_Float16*)take(4 * NE_ * 2);     // [B,512,32,32] f16
  float*    x1      = (float*)take(NE_ * 4);
  float*    x2      = (float*)take(NE_ * 4);
  float*    x3      = (float*)take(NE_ * 4);
  float*    xc      = (float*)take(NE_ * 4);
  _Float16* wproj_p = (_Float16*)take((size_t)6 * 128 * 128 * 2);
  _Float16* w3_p    = (_Float16*)take((size_t)128 * 4608 * 2);
  _Float16* w5_p    = (_Float16*)take((size_t)128 * 12800 * 2);
  _Float16* w7_p    = (_Float16*)take((size_t)128 * 25088 * 2);
  _Float16* w1_p    = (_Float16*)take((size_t)128 * 512 * 2);
  float*    stats   = (float*)take((size_t)4 * B_ * 2 * sizeof(float));

  // Stage 0: activation converts + weight fragment-packing.
  cvt_f32_f16<<<4096, 256, 0, stream>>>(st,  st_h,  NE_);
  cvt_f32_f16<<<4096, 256, 0, stream>>>(phy, phy_h, NE_);
  for (int i = 0; i < 6; ++i)
    pack_a<<<64, 256, 0, stream>>>(wproj[i], wproj_p + (size_t)i * 128 * 128,
                                   128, (size_t)128 * 128);
  pack_a<<<2304, 256, 0, stream>>>(w3, w3_p, 4608,  (size_t)128 * 4608);
  pack_a<<<6400, 256, 0, stream>>>(w5, w5_p, 12800, (size_t)128 * 12800);
  pack_a<<<12544, 256, 0, stream>>>(w7, w7_p, 25088, (size_t)128 * 25088);
  pack_a<<<256,  256, 0, stream>>>(w1, w1_p, 512,   (size_t)128 * 512);

  // Stage 1: six 1x1 projections (WMMA GEMM).
  proj_gemm<<<dim3(32, B_, 6), 256, 0, stream>>>(st_h, phy_h, wproj_p, pb, qkv);

  // Stage 2: four attention variants -> concat (WMMA + LDS softmax).
  attn_kernel<<<dim3(B_ * C_, 4), 128, 0, stream>>>(qkv, st, phy, concat);

  // Stage 3: implicit-GEMM convolutions (WMMA, the FLOP bulk).
  conv_gemm<3, 1><<<dim3(32, B_), 256, 0, stream>>>(concat, w3_p, nullptr, x1);
  conv_gemm<5, 2><<<dim3(32, B_), 256, 0, stream>>>(concat, w5_p, nullptr, x2);
  conv_gemm<7, 3><<<dim3(32, B_), 256, 0, stream>>>(concat, w7_p, nullptr, x3);
  conv_gemm<1, 0><<<dim3(32, B_), 256, 0, stream>>>(concat, w1_p, b1,      xc);

  // Stage 4: per-batch LayerNorm stats.
  ln_stats<<<B_, 256, 0, stream>>>(x1, stats + 0 * B_ * 2);
  ln_stats<<<B_, 256, 0, stream>>>(x2, stats + 1 * B_ * 2);
  ln_stats<<<B_, 256, 0, stream>>>(x3, stats + 2 * B_ * 2);
  ln_stats<<<B_, 256, 0, stream>>>(xc, stats + 3 * B_ * 2);

  // Stage 5: fused LN + sigmoid-mean + residual-LN combine.
  final_combine<<<(unsigned)((NE_ + 255) / 256), 256, 0, stream>>>(
      x1, x2, x3, xc, stats, g3, be3, g5, be5, g7, be7, g1, be1, (float*)d_out);
}
